// DeepSeekV3TokenChoiceTopKRouter_44976897524354
// MI455X (gfx1250) — matrix operand order repaired and checked by reading, not compiled, not run on hardware
//
#include <hip/hip_runtime.h>
#include <hip/hip_bf16.h>
#include <math.h>

// Problem constants (from reference)
#define DIM          7168
#define NUM_EXPERTS  256
#define EPT          8        // experts per token
#define NUM_GROUPS   8
#define GSIZE        32       // experts per group
#define TOPK_GROUPS  4
#define NT           16384    // tokens
#define FLATN        (NT * EPT)   // 131072
#define MTILE        64       // tokens per block in the GEMM

typedef __attribute__((ext_vector_type(16))) __bf16 bf16x16;
typedef __attribute__((ext_vector_type(8)))  float  f32x8;

__device__ __forceinline__ unsigned short f32_bf16(float f) {
  unsigned int u = __float_as_uint(f);
  u += 0x7FFFu + ((u >> 16) & 1u);   // round-to-nearest-even
  return (unsigned short)(u >> 16);
}

union BF16Frag { bf16x16 v; uint4 q[2]; };

// ---------------------------------------------------------------------------
// Phase 0b: one-time gate fp32 -> bf16 (3.7 MB, stays L2-resident afterwards)
// ---------------------------------------------------------------------------
__global__ __launch_bounds__(256) void convert_gate_kernel(
    const float* __restrict__ gate, unsigned int* __restrict__ gate_bf)
{
  const int i = blockIdx.x * blockDim.x + threadIdx.x;   // over E*DIM/2 pairs
  float2 f = ((const float2*)gate)[i];
  gate_bf[i] = (unsigned)f32_bf16(f.x) | ((unsigned)f32_bf16(f.y) << 16);
}

// ---------------------------------------------------------------------------
// Phase 1: logits[t][e] = dot(x[t], gate[e])  via bf16 WMMA, f32 accumulate.
// Block: 256 threads = 8 waves; M-tile = 64 tokens; wave w -> experts [32w,32w+32).
// Register blocking: per 32-wide K-step, B loaded once (2 N-tiles) and reused
// across 4 A-tiles -> 8 v_wmma per 8 ds_load_b128 + 4 global b128.
// ---------------------------------------------------------------------------
__global__ __launch_bounds__(256) void gemm_logits_kernel(
    const float* __restrict__ x, const unsigned short* __restrict__ gate_bf,
    float* __restrict__ logits)
{
  __shared__ __align__(16) unsigned short xs[MTILE][72];  // bf16 x-tile, padded

  const int tid   = threadIdx.x;
  const int wave  = tid >> 5;
  const int lane  = tid & 31;
  const int mbase = blockIdx.x * MTILE;

  // B layout (32x16 bf16): lane c<16 -> col c, K 0..15 ; lane c>=16 -> col c-16, K 16..31
  const int ecol  = lane & 15;
  const int khalf = (lane >> 4) << 4;          // 0 or 16 (elements)
  const int nbase = wave * 32;
  const unsigned short* gb0 = gate_bf + (size_t)(nbase +      ecol) * DIM + khalf;
  const unsigned short* gb1 = gate_bf + (size_t)(nbase + 16 + ecol) * DIM + khalf;

  // A layout (16x32 bf16): lane m<16 -> row m, K 0..15; lane m+16 -> K 16..31
  const int arow = lane & 15;
  const int aoff = (lane >> 4) << 4;           // 0 or 16

  f32x8 acc[4][2] = {};

  for (int kb = 0; kb < DIM; kb += 64) {
    // Stage x[64][kb..kb+64) as bf16 in LDS: 2048 packed pairs / 256 threads
#pragma unroll
    for (int i = 0; i < 8; ++i) {
      int p   = tid + (i << 8);
      int row = p >> 5;
      int col = (p & 31) << 1;
      float2 f = *(const float2*)&x[(size_t)(mbase + row) * DIM + kb + col];
      unsigned u = (unsigned)f32_bf16(f.x) | ((unsigned)f32_bf16(f.y) << 16);
      *(unsigned*)&xs[row][col] = u;
    }
    __syncthreads();

    if (kb + 64 < DIM) {                        // global_prefetch_b8 next gate chunk
      __builtin_prefetch(gb0 + kb + 64, 0, 1);
      __builtin_prefetch(gb1 + kb + 64, 0, 1);
    }

#pragma unroll
    for (int ks = 0; ks < 2; ++ks) {
      // B fragments: raw bf16, no conversion VALU
      const uint4* pb0 = (const uint4*)(gb0 + kb + ks * 32);
      const uint4* pb1 = (const uint4*)(gb1 + kb + ks * 32);
      BF16Frag B0, B1;
      B0.q[0] = pb0[0]; B0.q[1] = pb0[1];
      B1.q[0] = pb1[0]; B1.q[1] = pb1[1];

#pragma unroll
      for (int mt = 0; mt < 4; ++mt) {
        BF16Frag A;
        A.q[0] = *(const uint4*)&xs[mt * 16 + arow][ks * 32 + aoff];
        A.q[1] = *(const uint4*)&xs[mt * 16 + arow][ks * 32 + aoff + 8];
        acc[mt][0] = __builtin_amdgcn_wmma_f32_16x16x32_bf16(
            false, A.v, false, B0.v, (short)0, acc[mt][0], false, false);
        acc[mt][1] = __builtin_amdgcn_wmma_f32_16x16x32_bf16(
            false, A.v, false, B1.v, (short)0, acc[mt][1], false, false);
      }
    }
    __syncthreads();
  }

  // C/D layout: lane<16 -> M = r, N = lane ; lane>=16 -> M = 8+r, N = lane-16
#pragma unroll
  for (int mt = 0; mt < 4; ++mt) {
    const int rbase = mbase + mt * 16 + ((lane >> 4) << 3);
#pragma unroll
    for (int r = 0; r < 8; ++r) {
      logits[(size_t)(rbase + r) * NUM_EXPERTS + nbase +      ecol] = acc[mt][0][r];
      logits[(size_t)(rbase + r) * NUM_EXPERTS + nbase + 16 + ecol] = acc[mt][1][r];
    }
  }
}

// ---------------------------------------------------------------------------
// Phase 2: per-token routing. One thread per token; biased sigmoid scores are
// written back in-place over logits (thread owns its row).
// ---------------------------------------------------------------------------
__global__ __launch_bounds__(256) void route_kernel(
    float* __restrict__ scores,           // in: logits, out: biased scores
    const float* __restrict__ bias,
    float* __restrict__ weights, int* __restrict__ eids, int* __restrict__ counts)
{
  __shared__ float sbias[NUM_EXPERTS];
  sbias[threadIdx.x] = bias[threadIdx.x];
  __syncthreads();

  const int t = blockIdx.x * blockDim.x + threadIdx.x;
  float* L = scores + (size_t)t * NUM_EXPERTS;

  // group score = sum of top-2 biased sigmoid scores per group of 32
  float gs[NUM_GROUPS];
#pragma unroll
  for (int g = 0; g < NUM_GROUPS; ++g) {
    float m1 = -__builtin_inff(), m2 = -__builtin_inff();
    for (int j = 0; j < GSIZE; ++j) {
      const int e = g * GSIZE + j;
      float s = 1.0f / (1.0f + __expf(-L[e])) + sbias[e];
      L[e] = s;
      if (s > m1) { m2 = m1; m1 = s; } else if (s > m2) { m2 = s; }
    }
    gs[g] = m1 + m2;
  }

  // top-4 groups (ties -> lowest index, matches jax.lax.top_k)
  unsigned gmask = 0u;
#pragma unroll
  for (int i = 0; i < TOPK_GROUPS; ++i) {
    float best = -__builtin_inff(); int bg = 0;
#pragma unroll
    for (int g = 0; g < NUM_GROUPS; ++g)
      if (!((gmask >> g) & 1u) && gs[g] > best) { best = gs[g]; bg = g; }
    gmask |= 1u << bg;
  }

  // top-8 experts among allowed groups
  unsigned chosen[NUM_EXPERTS / 32] = {0, 0, 0, 0, 0, 0, 0, 0};
  int sel[EPT]; float us[EPT]; float sum = 0.0f;
  for (int k = 0; k < EPT; ++k) {
    float best = -__builtin_inff(); int be = 0;
    for (int e = 0; e < NUM_EXPERTS; ++e) {
      if (!((gmask >> (e >> 5)) & 1u)) { e += 31; continue; }   // skip whole group
      if ((chosen[e >> 5] >> (e & 31)) & 1u) continue;
      float s = L[e];
      if (s > best) { best = s; be = e; }
    }
    chosen[be >> 5] |= 1u << (be & 31);
    sel[k] = be;
    float u = L[be] - sbias[be];       // unbiased sigmoid score
    us[k] = u; sum += u;
  }

  const float scale = 2.5f / (sum + 1e-20f);
#pragma unroll
  for (int k = 0; k < EPT; ++k) {
    weights[t * EPT + k] = us[k] * scale;
    eids[t * EPT + k]    = sel[k];
    atomicAdd(&counts[sel[k]], 1);
  }
}

// ---------------------------------------------------------------------------
// Phase 0/3: counts zeroing and exclusive scan (single block of 256)
// ---------------------------------------------------------------------------
__global__ void zero_counts_kernel(int* __restrict__ counts) {
  counts[threadIdx.x] = 0;
}

__global__ __launch_bounds__(256) void scan_kernel(
    const int* __restrict__ counts, int* __restrict__ offsets,
    float* __restrict__ out_counts)
{
  __shared__ int tmp[NUM_EXPERTS];
  const int i = threadIdx.x;
  const int c = counts[i];
  out_counts[i] = (float)c;          // output 3: num_tokens_per_expert
  tmp[i] = c;
  __syncthreads();
#pragma unroll
  for (int off = 1; off < NUM_EXPERTS; off <<= 1) {
    int add = (i >= off) ? tmp[i - off] : 0;
    __syncthreads();
    tmp[i] += add;
    __syncthreads();
  }
  offsets[i] = tmp[i] - c;           // exclusive prefix
}

// ---------------------------------------------------------------------------
// Phase 4: stable counting-sort scatter. One wave per expert scans all flat
// assignments in order; ballot+popc gives the stable in-expert rank.
// ---------------------------------------------------------------------------
__global__ __launch_bounds__(32) void scatter_kernel(
    const int* __restrict__ eids, const float* __restrict__ weights,
    const int* __restrict__ offsets,
    float* __restrict__ out_scores, float* __restrict__ out_tok)
{
  const int e    = blockIdx.x;
  const int lane = threadIdx.x;
  int pos = offsets[e];
  for (int base = 0; base < FLATN; base += 32) {
    const int id = eids[base + lane];
    const bool hit = (id == e);
    const unsigned mask = __builtin_amdgcn_ballot_w32(hit);
    if (hit) {
      const int rank = __builtin_popcount(mask & ((1u << lane) - 1u));
      out_scores[pos + rank] = weights[base + lane];
      out_tok[pos + rank]    = (float)((base + lane) >> 3);   // flat / EPT
    }
    pos += __builtin_popcount(mask);
  }
}

// ---------------------------------------------------------------------------
extern "C" void kernel_launch(void* const* d_in, const int* in_sizes, int n_in,
                              void* d_out, int out_size, void* d_ws, size_t ws_size,
                              hipStream_t stream) {
  const float* x    = (const float*)d_in[0];   // [16384, 7168]
  const float* gate = (const float*)d_in[1];   // [256, 7168]
  const float* bias = (const float*)d_in[2];   // [256]

  float* out        = (float*)d_out;
  float* out_scores = out;                     // [131072]
  float* out_tok    = out + FLATN;             // [131072]
  float* out_cnt    = out + 2 * FLATN;         // [256]

  // workspace carve (~21.5 MB)
  char* ws = (char*)d_ws;
  size_t off = 0;
  float* logits  = (float*)(ws + off); off += (size_t)NT * NUM_EXPERTS * 4;      // 16 MB
  float* weights = (float*)(ws + off); off += (size_t)FLATN * 4;                 // 512 KB
  int*   eids    = (int*)  (ws + off); off += (size_t)FLATN * 4;                 // 512 KB
  unsigned short* gate_bf = (unsigned short*)(ws + off);
  off += (size_t)NUM_EXPERTS * DIM * 2;                                          // 3.7 MB
  int*   counts  = (int*)  (ws + off); off += NUM_EXPERTS * 4;
  int*   offsets = (int*)  (ws + off);

  zero_counts_kernel<<<1, NUM_EXPERTS, 0, stream>>>(counts);
  convert_gate_kernel<<<(NUM_EXPERTS * DIM / 2) / 256, 256, 0, stream>>>(
      gate, (unsigned int*)gate_bf);
  gemm_logits_kernel<<<NT / MTILE, 256, 0, stream>>>(x, gate_bf, logits);
  route_kernel<<<NT / 256, 256, 0, stream>>>(logits, bias, weights, eids, counts);
  scan_kernel<<<1, NUM_EXPERTS, 0, stream>>>(counts, offsets, out_cnt);
  scatter_kernel<<<NUM_EXPERTS, 32, 0, stream>>>(eids, weights, offsets,
                                                 out_scores, out_tok);
}
